// Block_15247133900803
// MI455X (gfx1250) — matrix operand order repaired and checked by reading, not compiled, not run on hardware
//
#include <hip/hip_runtime.h>

// ---------------------------------------------------------------------------
// Transformer block for MI455X (gfx1250, wave32, WMMA 16x16x32 f16->f32).
// All GEMMs + attention matmuls go through v_wmma_f32_16x16x32_f16.
// GEMM uses 64x64 output tiles per wave (16 WMMAs per 8KB of fragment loads)
// to lift cache-side arithmetic intensity to ~32 FLOP/byte.
// ---------------------------------------------------------------------------

#define TN   256                 // sequence length T
#define CE   384                 // n_embd
#define HS   64                  // head size
#define NH   6                   // heads
#define BS   128                 // batch
#define MTOK (BS * TN)           // 32768 token rows
#define FFD  (4 * CE)            // 1536

typedef __attribute__((ext_vector_type(16))) _Float16 v16h;
typedef __attribute__((ext_vector_type(8)))  _Float16 v8h;
typedef __attribute__((ext_vector_type(8)))  float    v8f;

// ---- WMMA fragment loaders ------------------------------------------------
// A (16x32 f16, row-major source, leading dim ld):
//   lane L: row = L&15 ; K-halves [kb..kb+7] and [kb+16..kb+23], kb = (L<16?0:8)
__device__ __forceinline__ v16h load_a_frag(const _Float16* base, int ld) {
  const int lane = threadIdx.x & 31;
  const _Float16* p = base + (size_t)(lane & 15) * ld + ((lane >> 4) << 3);
  union { v16h v; v8h h[2]; } u;
  u.h[0] = *(const v8h*)(p);
  u.h[1] = *(const v8h*)(p + 16);
  return u.v;
}

// B (32x16 f16) from a transposed (N x K row-major) source, leading dim ld:
//   lane L: col = L&15 ; K-halves [kb..kb+7] and [kb+16..kb+23], kb = (L<16?0:8)
__device__ __forceinline__ v16h load_b_frag(const _Float16* base, int ld) {
  const int lane = threadIdx.x & 31;
  const _Float16* p = base + (size_t)(lane & 15) * ld + ((lane >> 4) << 3);
  union { v16h v; v8h h[2]; } u;
  u.h[0] = *(const v8h*)(p);
  u.h[1] = *(const v8h*)(p + 16);
  return u.v;
}

// ---- generic f16 GEMM: out = A(MxK) @ BT(NxK)^T (+bias) -------------------
// Per-wave 64x64 tile: 4 A-frags x 4 B-frags -> 16 WMMAs per K-step.
// mode 0: store f32   mode 1: store f16   mode 2: store f16 relu
// mode 3: store f16 scattered as V^T [B,NH,HS,TN]
__global__ void __launch_bounds__(256) gemm_wmma(
    const _Float16* __restrict__ A, const _Float16* __restrict__ BT,
    const float* __restrict__ bias, void* __restrict__ outp,
    int M, int N, int K, int mode) {
  const int wave   = blockIdx.x * (blockDim.x >> 5) + (threadIdx.x >> 5);
  const int mtiles = M >> 6;                 // 64-row tiles
  const int nstrip = wave / mtiles;
  const int mt     = wave - nstrip * mtiles;
  const int m0 = mt << 6, n0 = nstrip << 6;
  if (n0 >= N) return;
  const int lane = threadIdx.x & 31;

  v8f z = {};
  v8f acc[16];
  for (int i = 0; i < 16; ++i) acc[i] = z;

  const _Float16* Ab = A  + (size_t)m0 * K;
  const _Float16* Bb = BT + (size_t)n0 * K;

  for (int k0 = 0; k0 < K; k0 += 32) {
    v16h a[4];
#pragma unroll
    for (int mi = 0; mi < 4; ++mi)
      a[mi] = load_a_frag(Ab + (size_t)(mi << 4) * K + k0, K);
#pragma unroll
    for (int t = 0; t < 4; ++t) {
      v16h b = load_b_frag(Bb + (size_t)(t << 4) * K + k0, K);
#pragma unroll
      for (int mi = 0; mi < 4; ++mi)
        acc[mi * 4 + t] = __builtin_amdgcn_wmma_f32_16x16x32_f16(
            false, a[mi], false, b, (short)0, acc[mi * 4 + t], false, false);
    }
  }

  const int col   = lane & 15;
  const int rhalf = (lane >> 4) << 3;        // C layout: rows v + {0,8}
  for (int mi = 0; mi < 4; ++mi) {
    for (int t = 0; t < 4; ++t) {
      const int n = n0 + (t << 4) + col;
      const float bv = bias ? bias[n] : 0.0f;
      for (int v = 0; v < 8; ++v) {
        float val = acc[mi * 4 + t][v] + bv;
        const int row = m0 + (mi << 4) + rhalf + v;
        if (mode == 0) {
          ((float*)outp)[(size_t)row * N + n] = val;
        } else if (mode == 1) {
          ((_Float16*)outp)[(size_t)row * N + n] = (_Float16)val;
        } else if (mode == 2) {
          ((_Float16*)outp)[(size_t)row * N + n] = (_Float16)(val > 0.f ? val : 0.f);
        } else { // V^T scatter: [B, NH, HS, TN]
          const int bb = row >> 8, tt = row & 255;
          const int hh = n >> 6,  dd = n & 63;
          ((_Float16*)outp)[((((size_t)bb * NH + hh) << 6) + dd) * TN + tt] = (_Float16)val;
        }
      }
    }
  }
}

// ---- flash attention: one wave per (b, h, 16-query block) ------------------
__global__ void __launch_bounds__(256) attn_wmma(
    const _Float16* __restrict__ Q, const _Float16* __restrict__ Kh,
    const _Float16* __restrict__ VT, _Float16* __restrict__ Ah) {
  __shared__ _Float16 pbuf[8][16][34];         // per-wave P transpose staging
  const int wid  = threadIdx.x >> 5;
  const int lane = threadIdx.x & 31;
  const int gw   = blockIdx.x * 8 + wid;
  const int qblk = gw & 15;
  const int bh   = gw >> 4;
  const int b = bh / NH, h = bh % NH;
  const int q0 = qblk << 4;

  const _Float16* Qb = Q + ((size_t)(b * TN + q0) * CE) + h * HS;
  v16h aq0 = load_a_frag(Qb, CE);              // dims 0..31
  v16h aq1 = load_a_frag(Qb + 32, CE);         // dims 32..63

  v8f z = {};
  v8f O[4] = {z, z, z, z};
  float mrow[8], lrow[8];
  for (int v = 0; v < 8; ++v) { mrow[v] = -1e30f; lrow[v] = 0.f; }

  const int col   = lane & 15;
  const int rhalf = (lane >> 4) << 3;
  const int kmax  = q0 + 16;                   // causal: keys < kmax

  for (int kb = 0; kb < kmax; kb += 32) {
    const _Float16* Kb = Kh + ((size_t)(b * TN + kb) * CE) + h * HS;
    v8f s0 = z, s1 = z;
    s0 = __builtin_amdgcn_wmma_f32_16x16x32_f16(false, aq0, false, load_b_frag(Kb, CE),      (short)0, s0, false, false);
    s0 = __builtin_amdgcn_wmma_f32_16x16x32_f16(false, aq1, false, load_b_frag(Kb + 32, CE), (short)0, s0, false, false);
    const bool hi = (kb + 16) < kmax;
    if (hi) {
      const _Float16* Kb2 = Kb + (size_t)16 * CE;
      s1 = __builtin_amdgcn_wmma_f32_16x16x32_f16(false, aq0, false, load_b_frag(Kb2, CE),      (short)0, s1, false, false);
      s1 = __builtin_amdgcn_wmma_f32_16x16x32_f16(false, aq1, false, load_b_frag(Kb2 + 32, CE), (short)0, s1, false, false);
    }

    float scs[8];
    for (int v = 0; v < 8; ++v) {
      const int qrow = q0 + rhalf + v;
      float x0 = s0[v] * 0.125f;               // 1/sqrt(64)
      if (kb + col > qrow) x0 = -1e30f;
      float x1 = hi ? s1[v] * 0.125f : -1e30f;
      if (kb + 16 + col > qrow) x1 = -1e30f;

      float mt = fmaxf(x0, x1);
      for (int off = 1; off < 16; off <<= 1) mt = fmaxf(mt, __shfl_xor(mt, off, 32));
      const float mnew = fmaxf(mrow[v], mt);
      const float sc = __expf(mrow[v] - mnew);
      const float p0 = __expf(x0 - mnew);
      const float p1 = __expf(x1 - mnew);
      float rs = p0 + p1;
      for (int off = 1; off < 16; off <<= 1) rs += __shfl_xor(rs, off, 32);
      lrow[v] = lrow[v] * sc + rs;
      mrow[v] = mnew;
      scs[v]  = sc;
      pbuf[wid][rhalf + v][col]      = (_Float16)p0;
      pbuf[wid][rhalf + v][16 + col] = (_Float16)p1;
    }
    for (int t = 0; t < 4; ++t)
      for (int v = 0; v < 8; ++v) O[t][v] *= scs[v];

    // wave-internal C-layout -> A-layout transpose through LDS
    asm volatile("s_wait_dscnt 0x0" ::: "memory");
    union { v16h v; v8h h[2]; } ap;
    {
      const int ar = lane & 15;
      const int ak = (lane < 16) ? 0 : 8;
      for (int i = 0; i < 8; ++i) {
        ap.h[0][i] = pbuf[wid][ar][ak + i];
        ap.h[1][i] = pbuf[wid][ar][ak + 16 + i];
      }
    }
    const _Float16* Vb = VT + ((size_t)bh << 6) * TN + kb;   // V^T rows = dims
    for (int t = 0; t < 4; ++t) {
      v16h bv = load_b_frag(Vb + (size_t)(t << 4) * TN, TN);
      O[t] = __builtin_amdgcn_wmma_f32_16x16x32_f16(false, ap.v, false, bv, (short)0, O[t], false, false);
    }
  }

  float inv[8];
  for (int v = 0; v < 8; ++v) inv[v] = 1.0f / lrow[v];
  for (int t = 0; t < 4; ++t)
    for (int v = 0; v < 8; ++v) {
      const int row = q0 + rhalf + v;
      const int c   = h * HS + (t << 4) + col;
      Ah[((size_t)(b * TN + row)) * CE + c] = (_Float16)(O[t][v] * inv[v]);
    }
}

// ---- fused LayerNorm + residual (fp32), one block (128 thr) per token ------
__global__ void __launch_bounds__(128) ln_residual(
    const float* __restrict__ s, const float* __restrict__ resid,
    const float* __restrict__ g, const float* __restrict__ beta,
    float* __restrict__ of32, _Float16* __restrict__ of16) {
  __shared__ float rs[128], rq[128];
  const int t = blockIdx.x, tid = threadIdx.x;
  const float* row = s + (size_t)t * CE;
  float a0 = row[tid], a1 = row[tid + 128], a2 = row[tid + 256];
  rs[tid] = a0 + a1 + a2;
  rq[tid] = a0 * a0 + a1 * a1 + a2 * a2;
  __syncthreads();
  for (int off = 64; off > 0; off >>= 1) {
    if (tid < off) { rs[tid] += rs[tid + off]; rq[tid] += rq[tid + off]; }
    __syncthreads();
  }
  const float mu   = rs[0] * (1.0f / CE);
  const float var  = rq[0] * (1.0f / CE) - mu * mu;
  const float rstd = rsqrtf(var + 1e-5f);
  const float* xr = resid + (size_t)t * CE;
  for (int i = 0; i < 3; ++i) {
    const int c = tid + i * 128;
    const float a = (i == 0) ? a0 : (i == 1) ? a1 : a2;
    const float y = xr[c] + (a - mu) * rstd * g[c] + beta[c];
    if (of32) of32[(size_t)t * CE + c] = y;
    if (of16) of16[(size_t)t * CE + c] = (_Float16)y;
  }
}

// ---- casts -----------------------------------------------------------------
__global__ void __launch_bounds__(256) cast_f32_f16(const float* __restrict__ in,
                                                    _Float16* __restrict__ out, int n) {
  const int i = blockIdx.x * 256 + threadIdx.x;
  if (i < n) out[i] = (_Float16)in[i];
}

// BT[n*K + k] = W[k*N + n]   (W is K x N row-major f32)
__global__ void __launch_bounds__(256) transpose_cast(const float* __restrict__ W,
                                                      _Float16* __restrict__ BT,
                                                      int K, int N) {
  const int idx = blockIdx.x * 256 + threadIdx.x;
  if (idx < N * K) {
    const int n = idx / K, k = idx - n * K;
    BT[idx] = (_Float16)W[(size_t)k * N + n];
  }
}

// stacked head weights [NH, CE, HS] -> BT[(h*64+d)*CE + k]
__global__ void __launch_bounds__(256) qkvw_cast(const float* __restrict__ W,
                                                 _Float16* __restrict__ BT) {
  const int idx = blockIdx.x * 256 + threadIdx.x;  // over CE*CE
  if (idx < CE * CE) {
    const int n = idx / CE, k = idx - n * CE;
    const int h = n >> 6, d = n & 63;
    BT[idx] = (_Float16)W[((size_t)h * CE + k) * HS + d];
  }
}

// ---------------------------------------------------------------------------
extern "C" void kernel_launch(void* const* d_in, const int* in_sizes, int n_in,
                              void* d_out, int out_size, void* d_ws, size_t ws_size,
                              hipStream_t stream) {
  (void)in_sizes; (void)n_in; (void)out_size; (void)ws_size;
  const float* x   = (const float*)d_in[0];
  const float* Wq  = (const float*)d_in[1];
  const float* Wk  = (const float*)d_in[2];
  const float* Wv  = (const float*)d_in[3];
  const float* Wp  = (const float*)d_in[4];
  const float* bp  = (const float*)d_in[5];
  const float* W1  = (const float*)d_in[6];
  const float* b1  = (const float*)d_in[7];
  const float* W2  = (const float*)d_in[8];
  const float* b2  = (const float*)d_in[9];
  const float* g1  = (const float*)d_in[10];
  const float* be1 = (const float*)d_in[11];
  const float* g2  = (const float*)d_in[12];
  const float* be2 = (const float*)d_in[13];

  // workspace layout (bytes); aliases noted. total ~255 MB
  char* ws = (char*)d_ws;
  _Float16* xh  = (_Float16*)(ws + 0);            // 25,165,824 B
  _Float16* qh  = (_Float16*)(ws + 25165824);     // 25,165,824 B
  _Float16* kh  = (_Float16*)(ws + 50331648);     // 25,165,824 B
  _Float16* vth = (_Float16*)(ws + 75497472);     // 25,165,824 B  (V^T)
  _Float16* ath = (_Float16*)(ws + 100663296);    // 25,165,824 B  (attn out)
  _Float16* h1h = qh;                             // FFN hidden aliases qh..ath (100 MB)
  float*    sa  = (float*)(ws + 125829120);       // 50,331,648 B
  float*    ff  = sa;                             // aliases sa (sa dead after ln1)
  float*    x1  = (float*)(ws + 176160768);       // 50,331,648 B
  _Float16* x1h = (_Float16*)(ws + 226492416);    // 25,165,824 B
  _Float16* wqT = (_Float16*)(ws + 251658240);
  _Float16* wkT = wqT + CE * CE;
  _Float16* wvT = wkT + CE * CE;
  _Float16* wpT = wvT + CE * CE;
  _Float16* w1T = wpT + CE * CE;                  // [FFD][CE]
  _Float16* w2T = w1T + (size_t)FFD * CE;         // [CE][FFD]

  // 1) casts / weight transposes
  cast_f32_f16<<<(MTOK * CE) / 256, 256, 0, stream>>>(x, xh, MTOK * CE);
  qkvw_cast<<<(CE * CE) / 256, 256, 0, stream>>>(Wq, wqT);
  qkvw_cast<<<(CE * CE) / 256, 256, 0, stream>>>(Wk, wkT);
  qkvw_cast<<<(CE * CE) / 256, 256, 0, stream>>>(Wv, wvT);
  transpose_cast<<<(CE * CE) / 256, 256, 0, stream>>>(Wp, wpT, CE, CE);
  transpose_cast<<<(CE * FFD) / 256, 256, 0, stream>>>(W1, w1T, CE, FFD);
  transpose_cast<<<(CE * FFD) / 256, 256, 0, stream>>>(W2, w2T, FFD, CE);

  // 2) QKV projections (V written directly transposed per head)
  const int qkv_blocks = ((MTOK / 64) * (CE / 64)) / 8;    // 384
  gemm_wmma<<<qkv_blocks, 256, 0, stream>>>(xh, wqT, nullptr, qh,  MTOK, CE, CE, 1);
  gemm_wmma<<<qkv_blocks, 256, 0, stream>>>(xh, wkT, nullptr, kh,  MTOK, CE, CE, 1);
  gemm_wmma<<<qkv_blocks, 256, 0, stream>>>(xh, wvT, nullptr, vth, MTOK, CE, CE, 3);

  // 3) causal flash attention
  attn_wmma<<<(BS * NH * (TN / 16)) / 8, 256, 0, stream>>>(qh, kh, vth, ath);

  // 4) output projection + LN1 + residual
  gemm_wmma<<<qkv_blocks, 256, 0, stream>>>(ath, wpT, bp, sa, MTOK, CE, CE, 0);
  ln_residual<<<MTOK, 128, 0, stream>>>(sa, x, g1, be1, x1, x1h);

  // 5) FFN
  const int ffn1_blocks = ((MTOK / 64) * (FFD / 64)) / 8;  // 1536
  gemm_wmma<<<ffn1_blocks, 256, 0, stream>>>(x1h, w1T, b1, h1h, MTOK, FFD, CE, 2);
  gemm_wmma<<<qkv_blocks, 256, 0, stream>>>(h1h, w2T, b2, ff, MTOK, CE, FFD, 0);

  // 6) LN2 + residual -> final output
  ln_residual<<<MTOK, 128, 0, stream>>>(ff, x1, g2, be2, (float*)d_out, nullptr);
}
